// MACEInteractionBlock_63977832841984
// MI455X (gfx1250) — matrix operand order repaired and checked by reading, not compiled, not run on hardware
//
#include <hip/hip_runtime.h>
#include <hip/hip_bf16.h>
#include <math.h>

// ---------------- WMMA types & helpers (CDNA5 gfx1250, wave32) -------------
typedef __attribute__((ext_vector_type(16))) __bf16 v16bf;
typedef __attribute__((ext_vector_type(8)))  __bf16 v8bf;
typedef __attribute__((ext_vector_type(8)))  float  v8f;
typedef __attribute__((ext_vector_type(4)))  unsigned int u32x4;
typedef __attribute__((ext_vector_type(8)))  int i32x8;
typedef __attribute__((ext_vector_type(4)))  int i32x4;

__device__ inline v8f wmma_bf16(v16bf a, v16bf b, v8f c) {
  // D = A(16x32 bf16) * B(32x16 bf16) + C(16x16 f32)
  return __builtin_amdgcn_wmma_f32_16x16x32_bf16(
      false, a, false, b, (short)0, c, false, false);
}

// A-matrix (16x32, 16-bit) per-lane element -> K index.
__device__ inline int a_k(int lane, int j) {
  int khalf = (lane >> 4) << 3;                    // 0 or 8
  return (j < 8) ? (khalf + j) : (khalf + 8 + j);  // j>=8: 16+khalf+(j-8)
}
// B-matrix (32x16, 16-bit) per-lane element -> K index.
__device__ inline int b_k(int lane, int j) { return ((lane >> 4) << 3) + ((j < 8) ? j : (j + 8)); }
// C/D (16x16 f32): lane holds column (lane&15), rows ((lane>>4)*8 + i)
__device__ inline int c_row(int lane, int i) { return ((lane >> 4) << 3) + i; }

// A fragment from a bf16 row-major tile (row stride `stride` elems):
// per lane: 8 contiguous bf16 at K=kk+khalf, 8 more at K=kk+khalf+16.
// Each half is 16B => one ds_load_b128 / global_load_b128.
__device__ inline v16bf load_a_frag(const __bf16* base, int lane, int stride,
                                    int kk) {
  const __bf16* row =
      base + (size_t)(lane & 15) * stride + kk + ((lane >> 4) << 3);
  v8bf lo = *(const v8bf*)(row);
  v8bf hi = *(const v8bf*)(row + 16);
  v16bf a;
#pragma unroll
  for (int j = 0; j < 8; ++j) { a[j] = lo[j]; a[8 + j] = hi[j]; }
  return a;
}
// B fragment from the pre-packed weight buffer (tile = 32 lanes x 16 bf16).
__device__ inline v16bf load_b_frag(const __bf16* packed, int tile, int lane) {
  return *(const v16bf*)(packed + (size_t)tile * 512 + lane * 16);
}

#define N_NODES 8192
#define N_EDGES 131072
#define MUL 128
#define SH_DIM 16

static __device__ __constant__ float INV_MUL = 0.08838834764831845f; // 1/sqrt(128)
static __device__ __constant__ float INV_RB  = 0.35355339059327373f; // 1/sqrt(8)
static __device__ __constant__ float INV_RM  = 0.125f;               // 1/sqrt(64)

__device__ inline float silu(float x) { return x / (1.f + __expf(-x)); }

// ------------------------- Tensor Data Mover helpers -----------------------
// 1-D contiguous tile load Global -> LDS via TDM (D# per ISA 08_async_tensor).
__device__ inline void tdm_load_1d(unsigned lds_off, unsigned long long gaddr,
                                   unsigned nelem, unsigned dsz_code) {
#if defined(__gfx1250__) && __has_builtin(__builtin_amdgcn_tensor_load_to_lds)
  u32x4 g0;
  g0[0] = 1u;                                        // count=1, user D#
  g0[1] = lds_off;                                   // lds_addr (bytes)
  g0[2] = (unsigned)(gaddr & 0xFFFFFFFFull);         // global_addr[31:0]
  g0[3] = (unsigned)((gaddr >> 32) & 0x1FFFFFFull)   // global_addr[56:32]
          | (2u << 30);                              // type=2 ("image")
  i32x8 g1;
  g1[0] = (int)(dsz_code << 16);                     // data_size
  g1[1] = (int)((nelem & 0xFFFFu) << 16);            // tensor_dim0[15:0]
  g1[2] = (int)(nelem >> 16);                        // tensor_dim0[31:16]
  g1[3] = (int)((nelem & 0xFFFFu) << 16);            // tile_dim0
  g1[4] = 0;                                         // tile_dim1/2 = 0
  g1[5] = (int)nelem;                                // tensor_dim0_stride lo
  g1[6] = 0;
  g1[7] = 0;
  i32x4 z4 = (i32x4)0;
#if __clang_major__ >= 23
  i32x8 z8 = (i32x8)0;
  __builtin_amdgcn_tensor_load_to_lds(g0, g1, z4, z4, z8, 0);
#else
  __builtin_amdgcn_tensor_load_to_lds(g0, g1, z4, z4, 0);
#endif
#endif
}
__device__ inline void tdm_wait() {
#if defined(__gfx1250__) && __has_builtin(__builtin_amdgcn_s_wait_tensorcnt)
  __builtin_amdgcn_s_wait_tensorcnt(0);
#endif
}
#if defined(__gfx1250__) && __has_builtin(__builtin_amdgcn_tensor_load_to_lds)
#define HAVE_TDM 1
#else
#define HAVE_TDM 0
#endif

// ---------------------------------------------------------------------------
// Kernel 0: zero the agg scratch [16][N_NODES][MUL] f32
// ---------------------------------------------------------------------------
__global__ void zero_kernel(float4* __restrict__ p, int n4) {
  int i = blockIdx.x * blockDim.x + threadIdx.x;
  int stride = gridDim.x * blockDim.x;
  float4 z; z.x = 0.f; z.y = 0.f; z.z = 0.f; z.w = 0.f;
  for (; i < n4; i += stride) p[i] = z;
}

// ---------------------------------------------------------------------------
// Kernel 0b: f32 -> bf16 bulk convert (node_feats staging)
// ---------------------------------------------------------------------------
__global__ void cvt_bf16_kernel(const float* __restrict__ src,
                                __bf16* __restrict__ dst, int n) {
  int i = blockIdx.x * blockDim.x + threadIdx.x;
  int stride = gridDim.x * blockDim.x;
  for (; i < n; i += stride) dst[i] = (__bf16)src[i];
}

// ---------------------------------------------------------------------------
// Kernel 0c: pack an f32 [srcK x N] weight (zero-padded in K to Ktiles*32)
// into bf16 WMMA-B fragment layout: tile (kt,nt) -> 32 lanes x 16 bf16,
// lane element j = W[kt*32 + b_k(lane,j)][nt*16 + (lane&15)].
// ---------------------------------------------------------------------------
__global__ __launch_bounds__(64) void pack_kernel(
    const float* __restrict__ src, int srcK, int N, int Ntiles,
    __bf16* __restrict__ dst) {
  const int kt = blockIdx.x / Ntiles;
  const int nt = blockIdx.x % Ntiles;
  const int lane = threadIdx.x & 31;
  const int half = threadIdx.x >> 5;
  const int n = nt * 16 + (lane & 15);
  __bf16* out = dst + (size_t)blockIdx.x * 512 + lane * 16 + half * 8;
#pragma unroll
  for (int j = 0; j < 8; ++j) {
    int k = kt * 32 + b_k(lane, half * 8 + j);
    out[j] = (__bf16)((k < srcK) ? src[(size_t)k * N + n] : 0.f);
  }
}

// ---------------------------------------------------------------------------
// Kernel 1: feat = node_feats @ W_up * (1/sqrt(MUL))
// A tile (16x128 bf16, contiguous 4KB) staged via TDM; B from packed wup.
// ---------------------------------------------------------------------------
__global__ __launch_bounds__(256) void feat_kernel(
    const __bf16* __restrict__ nf_bf, const __bf16* __restrict__ wup_p,
    float* __restrict__ feat) {
  const int warp = threadIdx.x >> 5;  // 0..7 -> d tile
  const int lane = threadIdx.x & 31;
  const int n0 = blockIdx.x * 16;

  __shared__ __align__(16) __bf16 tA[16 * 128];
#if HAVE_TDM
  if (threadIdx.x < 32) {
    tdm_load_1d((unsigned)(size_t)tA,
                (unsigned long long)(size_t)(nf_bf + (size_t)n0 * 128),
                2048u, /*bf16*/ 1u);
    tdm_wait();
  }
  __syncthreads();
#else
  for (int idx = threadIdx.x; idx < 2048; idx += blockDim.x)
    tA[idx] = nf_bf[(size_t)n0 * 128 + idx];
  __syncthreads();
#endif

  v8f c = {};
#pragma unroll
  for (int kt = 0; kt < 4; ++kt) {
    v16bf a = load_a_frag(tA, lane, 128, kt * 32);
    v16bf b = load_b_frag(wup_p, kt * 8 + warp, lane);
    c = wmma_bf16(a, b, c);
  }
  const int d = warp * 16 + (lane & 15);
#pragma unroll
  for (int i = 0; i < 8; ++i)
    feat[(size_t)(n0 + c_row(lane, i)) * 128 + d] = c[i] * INV_MUL;
}

// ---------------------------------------------------------------------------
// Kernel 2: sc0[n] = node_feats[n] . W_skip[species[n]] * inv ; zero sc tail
// ---------------------------------------------------------------------------
__global__ __launch_bounds__(128) void sc_kernel(
    const float* __restrict__ nf, const int* __restrict__ species,
    const float* __restrict__ Wskip, float* __restrict__ sc) {
  const int n = blockIdx.x;
  const int d = threadIdx.x;  // 0..127
  __shared__ float x[128];
  x[d] = nf[(size_t)n * 128 + d];
  __syncthreads();
  const float* W = Wskip + (size_t)species[n] * 128 * 128;
  float acc = 0.f;
#pragma unroll 8
  for (int c = 0; c < 128; ++c) acc = fmaf(x[c], W[(size_t)c * 128 + d], acc);
  float* row = sc + (size_t)n * 512;
  row[d] = acc * INV_MUL;
  row[128 + d] = 0.f;
  row[256 + d] = 0.f;
  row[384 + d] = 0.f;
}

// ---------------------------------------------------------------------------
// MLP middle stage: hout = silu(hin @ W * 1/8), 16x64 @ 64x64, bf16 LDS.
// A fragments hoisted (invariant across the 4 output tiles).
// ---------------------------------------------------------------------------
__device__ inline void mlp_stage(const __bf16* hin, const __bf16* Wp,
                                 __bf16* hout, int lane, float scale) {
  const int col = lane & 15;
  v16bf a0 = load_a_frag(hin, lane, 64, 0);
  v16bf a1 = load_a_frag(hin, lane, 64, 32);
#pragma unroll
  for (int t = 0; t < 4; ++t) {
    v8f c = {};
    c = wmma_bf16(a0, load_b_frag(Wp, 0 * 4 + t, lane), c);
    c = wmma_bf16(a1, load_b_frag(Wp, 1 * 4 + t, lane), c);
#pragma unroll
    for (int i = 0; i < 8; ++i)
      hout[c_row(lane, i) * 64 + t * 16 + col] = (__bf16)silu(c[i] * scale);
  }
}

// ---------------------------------------------------------------------------
// Kernel 3: edge kernel. 4 waves/block, 16 edges per wave.
// Radial MLP via WMMA (bf16 LDS activations, pre-packed weights). The wave's
// 16 sender feat rows and sh rows are staged into LDS once with b128 loads;
// the message outer-product is scattered straight from C fragments into
// agg[m][receiver][c] with global f32 atomics.
// ---------------------------------------------------------------------------
__global__ __launch_bounds__(128) void edge_kernel(
    const float* __restrict__ rb,   // [E,8]
    const float* __restrict__ sh,   // [E,16]
    const int*   __restrict__ snd, const int* __restrict__ rcv,
    const float* __restrict__ feat, // [N,128] f32
    const __bf16* __restrict__ wr0_p,   // 1x4 tiles (K padded 8->32)
    const __bf16* __restrict__ wr1_p,   // 2x4 tiles
    const __bf16* __restrict__ wr2_p,   // 2x4 tiles
    const __bf16* __restrict__ wr3_p,   // 2x32 tiles
    float* __restrict__ agg)            // [16][N,128]
{
  const int warp = threadIdx.x >> 5;  // 0..3
  const int lane = threadIdx.x & 31;
  const int e0 = (blockIdx.x * 4 + warp) * 16;
  const int col = lane & 15;

  __shared__ __align__(16) __bf16 hA[4][16 * 64];     //  8 KB
  __shared__ __align__(16) __bf16 hB[4][16 * 64];     //  8 KB
  __shared__ __align__(16) float  featL[4][16 * 128]; // 32 KB
  __shared__ __align__(16) float  shL[4][16 * 16];    //  4 KB
  __bf16* h_in = hA[warp];
  __bf16* h_out = hB[warp];

  // ---- stage the wave's 16 sender feat rows + sh rows into LDS (b128) ----
  {
    const int row = lane >> 1;              // 0..15
    const int halfo = (lane & 1) * 64;      // which half of the 128-row
    const int srow = snd[e0 + row];
    const float4* src = (const float4*)(feat + (size_t)srow * 128 + halfo);
    float4* dst = (float4*)(featL[warp] + row * 128 + halfo);
#pragma unroll
    for (int q = 0; q < 16; ++q) dst[q] = src[q];
    const float4* ssrc = (const float4*)(sh + (size_t)e0 * 16) + lane * 2;
    float4* sdst = (float4*)(shL[warp]) + lane * 2;
    sdst[0] = ssrc[0];
    sdst[1] = ssrc[1];
  }

  // ---- stage 1: h0 = silu(rb @ Wr0 * 1/sqrt(8)); K zero-padded 8->32 ----
  {
    v16bf a;
#pragma unroll
    for (int j = 0; j < 16; ++j) {
      int k = a_k(lane, j);
      a[j] = (__bf16)((k < 8) ? rb[(size_t)(e0 + col) * 8 + k] : 0.f);
    }
#pragma unroll
    for (int t = 0; t < 4; ++t) {
      v8f c = {};
      c = wmma_bf16(a, load_b_frag(wr0_p, t, lane), c);
#pragma unroll
      for (int i = 0; i < 8; ++i)
        h_in[c_row(lane, i) * 64 + t * 16 + col] = (__bf16)silu(c[i] * INV_RB);
    }
  }
  __syncthreads();

  // ---- stages 2,3 ----
  mlp_stage(h_in, wr1_p, h_out, lane, INV_RM);
  __syncthreads();
  mlp_stage(h_out, wr2_p, h_in, lane, INV_RM);
  __syncthreads();

  // ---- per-lane receiver ids ----
  int rreg[8];
#pragma unroll
  for (int i = 0; i < 8; ++i) rreg[i] = rcv[e0 + c_row(lane, i)];

  // ---- stage 4: rw = h2 @ Wr3 * 1/8, tile-by-tile, fused scatter ----
  v16bf a0 = load_a_frag(h_in, lane, 64, 0);
  v16bf a1 = load_a_frag(h_in, lane, 64, 32);
  const int m_lo[4] = {0, 1, 4, 9};
  const int m_hi[4] = {1, 4, 9, 16};
#pragma unroll 1
  for (int t = 0; t < 32; ++t) {
    const int l = t >> 3;               // L-channel block of 128
    const int cc = (t & 7) * 16 + col;  // channel within [0,128)
    v8f c = {};
    c = wmma_bf16(a0, load_b_frag(wr3_p, 0 * 32 + t, lane), c);
    c = wmma_bf16(a1, load_b_frag(wr3_p, 1 * 32 + t, lane), c);
#pragma unroll
    for (int i = 0; i < 8; ++i) {
      const int er = c_row(lane, i);                 // edge within tile
      const float w = c[i] * INV_RM;                 // rw[e][l*128+cc]
      const float fs = featL[warp][er * 128 + cc];   // msg_scal (LDS)
      const float base = w * fs;
      float* aggr = agg + (size_t)rreg[i] * 128 + cc;
      for (int m = m_lo[l]; m < m_hi[l]; ++m)
        atomicAdd(aggr + (size_t)m * N_NODES * 128,
                  base * shL[warp][er * 16 + m]);
    }
  }
}

// ---------------------------------------------------------------------------
// Kernel 4: out[:, d, m] = agg[m] @ W_lin[l(m)] * inv / 16
// A tile (16x128 f32, contiguous 8KB) staged via TDM, converted to bf16 LDS.
// ---------------------------------------------------------------------------
__global__ __launch_bounds__(256) void out_kernel(
    const float* __restrict__ agg, const __bf16* __restrict__ wlin_p,
    float* __restrict__ out) {
  const int m = blockIdx.y;  // 0..15 global sh component
  const int n0 = blockIdx.x * 16;
  const int warp = threadIdx.x >> 5;
  const int lane = threadIdx.x & 31;

  int l, ml, cb, dim;
  if (m == 0)      { l = 0; ml = 0;     cb = 0;    dim = 1; }
  else if (m < 4)  { l = 1; ml = m - 1; cb = 128;  dim = 3; }
  else if (m < 9)  { l = 2; ml = m - 4; cb = 512;  dim = 5; }
  else             { l = 3; ml = m - 9; cb = 1152; dim = 7; }

  const float* A = agg + (size_t)m * N_NODES * 128;

  __shared__ __align__(16) __bf16 tA[16 * 128];
#if HAVE_TDM
  __shared__ __align__(16) float tAf[16 * 128];
  if (threadIdx.x < 32) {
    tdm_load_1d((unsigned)(size_t)tAf,
                (unsigned long long)(size_t)(A + (size_t)n0 * 128),
                2048u, /*f32*/ 2u);
    tdm_wait();
  }
  __syncthreads();
  for (int idx = threadIdx.x; idx < 2048; idx += blockDim.x)
    tA[idx] = (__bf16)tAf[idx];
  __syncthreads();
#else
  for (int idx = threadIdx.x; idx < 2048; idx += blockDim.x)
    tA[idx] = (__bf16)A[(size_t)n0 * 128 + idx];
  __syncthreads();
#endif

  v8f c = {};
#pragma unroll
  for (int kt = 0; kt < 4; ++kt) {
    v16bf a = load_a_frag(tA, lane, 128, kt * 32);
    v16bf b = load_b_frag(wlin_p, (l * 4 + kt) * 8 + warp, lane);
    c = wmma_bf16(a, b, c);
  }
  const float scale = INV_MUL * (1.f / 16.f);  // * inv / AVG_NEIGH
  const int d = warp * 16 + (lane & 15);
#pragma unroll
  for (int i = 0; i < 8; ++i) {
    const int n = n0 + c_row(lane, i);
    out[(size_t)n * 2048 + cb + d * dim + ml] = c[i] * scale;
  }
}

// ---------------------------------------------------------------------------
extern "C" void kernel_launch(void* const* d_in, const int* in_sizes, int n_in,
                              void* d_out, int out_size, void* d_ws, size_t ws_size,
                              hipStream_t stream) {
  const float* node_feats = (const float*)d_in[0];
  const int*   species    = (const int*)d_in[1];
  const float* sh         = (const float*)d_in[2];
  const float* rb         = (const float*)d_in[3];
  const int*   senders    = (const int*)d_in[4];
  const int*   receivers  = (const int*)d_in[5];
  const float* Wup        = (const float*)d_in[6];
  const float* Wlin       = (const float*)d_in[7];
  const float* Wskip      = (const float*)d_in[8];
  const float* Wr0        = (const float*)d_in[9];
  const float* Wr1        = (const float*)d_in[10];
  const float* Wr2        = (const float*)d_in[11];
  const float* Wr3        = (const float*)d_in[12];

  float* message = (float*)d_out;                     // [8192,2048]
  float* sc      = message + (size_t)N_NODES * 2048;  // [8192,512]

  // workspace layout
  char* ws = (char*)d_ws;
  float*  agg   = (float*)ws;                                   // 64 MB
  float*  feat  = (float*)(ws + (size_t)SH_DIM * N_NODES * MUL * 4);
  __bf16* nf_bf = (__bf16*)((char*)feat + (size_t)N_NODES * MUL * 4);
  __bf16* wup_p  = nf_bf + (size_t)N_NODES * MUL;  // 32 tiles
  __bf16* wr0_p  = wup_p + 32 * 512;               // 4 tiles
  __bf16* wr1_p  = wr0_p + 4 * 512;                // 8 tiles
  __bf16* wr2_p  = wr1_p + 8 * 512;                // 8 tiles
  __bf16* wr3_p  = wr2_p + 8 * 512;                // 64 tiles
  __bf16* wlin_p = wr3_p + 64 * 512;               // 128 tiles

  const int agg_f4 = (SH_DIM * N_NODES * MUL) / 4;
  zero_kernel<<<1024, 256, 0, stream>>>((float4*)agg, agg_f4);
  cvt_bf16_kernel<<<512, 256, 0, stream>>>(node_feats, nf_bf, N_NODES * MUL);
  pack_kernel<<<32, 64, 0, stream>>>(Wup, 128, 128, 8, wup_p);
  pack_kernel<<<4,  64, 0, stream>>>(Wr0, 8,   64,  4, wr0_p);
  pack_kernel<<<8,  64, 0, stream>>>(Wr1, 64,  64,  4, wr1_p);
  pack_kernel<<<8,  64, 0, stream>>>(Wr2, 64,  64,  4, wr2_p);
  pack_kernel<<<64, 64, 0, stream>>>(Wr3, 64, 512, 32, wr3_p);
  for (int l = 0; l < 4; ++l)
    pack_kernel<<<32, 64, 0, stream>>>(Wlin + (size_t)l * 128 * 128, 128, 128,
                                       8, wlin_p + (size_t)l * 32 * 512);

  feat_kernel<<<N_NODES / 16, 256, 0, stream>>>(nf_bf, wup_p, feat);
  sc_kernel<<<N_NODES, 128, 0, stream>>>(node_feats, species, Wskip, sc);
  edge_kernel<<<N_EDGES / 64, 128, 0, stream>>>(rb, sh, senders, receivers,
                                                feat, wr0_p, wr1_p, wr2_p,
                                                wr3_p, agg);
  out_kernel<<<dim3(N_NODES / 16, 16), 256, 0, stream>>>(agg, wlin_p, message);
}
